// SelfAttentionWithGaussianBias_26774826123573
// MI455X (gfx1250) — matrix operand order repaired and checked by reading, not compiled, not run on hardware
//
#include <hip/hip_runtime.h>
#include <hip/hip_bf16.h>

typedef __attribute__((ext_vector_type(16))) __bf16 v16bf;
typedef __attribute__((ext_vector_type(8)))  __bf16 v8bf;
typedef __attribute__((ext_vector_type(8)))  float  v8f;
typedef __attribute__((ext_vector_type(4)))  unsigned int v4u;
typedef __attribute__((ext_vector_type(8)))  int v8i;
typedef __attribute__((ext_vector_type(4)))  int v4i;

#define B_   2
#define L_   2048
#define D_   512
#define H_   8
#define DH_  64
#define E3_  1536
#define M_   (B_ * L_)   // 4096
#define KPAD 72          // 64 elements + 4-DWORD TDM pad -> 144B row stride in LDS

// ---------------------------------------------------------------- fragments

__device__ __forceinline__ v16bf make16(v8bf lo, v8bf hi) {
    v16bf r;
#pragma unroll
    for (int i = 0; i < 8; ++i) { r[i] = lo[i]; r[i + 8] = hi[i]; }
    return r;
}

// A fragment (16x32 bf16, MxK) from row-major [m][k] storage.
__device__ __forceinline__ v16bf load_a_frag(const __bf16* __restrict__ base, int ld,
                                             int m0, int k0, int lane) {
    const int m = m0 + (lane & 15);
    const int g = lane >> 4;
    const __bf16* p = base + (size_t)m * ld + k0 + g * 16;
    v8bf lo = *(const v8bf*)(p);
    v8bf hi = *(const v8bf*)(p + 8);
    return make16(lo, hi);
}

// B fragment (32x16 bf16, KxN) from row-major [n][k] storage (32B-aligned rows).
__device__ __forceinline__ v16bf load_b_frag(const __bf16* __restrict__ base, int ld,
                                             int n0, int k0, int lane) {
    const int n = n0 + (lane & 15);
    const int g = lane >> 4;
    return *(const v16bf*)(base + (size_t)n * ld + k0 + g * 16);
}

// B fragment from LDS with 144B (16B-aligned, not 32B) row stride: two v8bf reads.
__device__ __forceinline__ v16bf load_b_frag_lds(const __bf16* base, int ld,
                                                 int n0, int k0, int lane) {
    const int n = n0 + (lane & 15);
    const int g = lane >> 4;
    const __bf16* p = base + n * ld + k0 + g * 16;
    v8bf lo = *(const v8bf*)(p);
    v8bf hi = *(const v8bf*)(p + 8);
    return make16(lo, hi);
}

// ---------------------------------------------------------------- TDM helper

// 2D tile load via Tensor Data Mover. data_size = 2 bytes (bf16).
// LDS padding: 4 DWORDs after every 32 DWORDs -> row stride KPAD elements.
__device__ __forceinline__ void tdm_load_2d(unsigned lds_off, const void* gptr,
                                            unsigned tensor_d0, unsigned tensor_d1,
                                            unsigned stride0,
                                            unsigned tile0, unsigned tile1) {
    const unsigned long long ga = (unsigned long long)gptr;
    v4u g0;
    g0[0] = 1u;                                            // count=1 (user descriptor)
    g0[1] = lds_off;                                       // lds_addr (bytes)
    g0[2] = (unsigned)(ga & 0xffffffffu);                  // global_addr lo
    g0[3] = (unsigned)((ga >> 32) & 0x01ffffffu) | (2u << 30); // global_addr hi | type=2
    v8i g1;
    g1[0] = (int)((1u << 16)      // data_size = 1 -> 2 bytes
                | (1u << 20)      // pad_enable
                | (4u << 22)      // pad_interval: every 32 DWORDs (one 64-elem row)
                | (3u << 25));    // pad_amount: 4 DWORDs
    g1[1] = (int)((tensor_d0 & 0xffffu) << 16);            // tensor_dim0[15:0]
    g1[2] = (int)((tensor_d0 >> 16) | ((tensor_d1 & 0xffffu) << 16));
    g1[3] = (int)((tensor_d1 >> 16) | ((tile0 & 0xffffu) << 16));
    g1[4] = (int)(tile1 & 0xffffu);                        // tile_dim1; tile_dim2 = 0
    g1[5] = (int)stride0;                                  // tensor_dim0_stride lo
    g1[6] = 0;                                             // stride0 hi | stride1 lo
    g1[7] = 0;
    v4i z4 = {0, 0, 0, 0};
    v8i z8 = {0, 0, 0, 0, 0, 0, 0, 0};
    __builtin_amdgcn_tensor_load_to_lds(g0, g1, z4, z4, z8, 0);
}

// ---------------------------------------------------------------- kernels

__global__ void cvt_f32_to_bf16_kernel(const float* __restrict__ in,
                                       __bf16* __restrict__ out, int n) {
    int i = blockIdx.x * blockDim.x + threadIdx.x;
    if (i < n) out[i] = (__bf16)in[i];
}

// qkv = x @ w_qkv^T ; M=4096, N=1536, K=512. Writes Q,K row-major, V transposed.
__global__ __launch_bounds__(256)
void qkv_gemm_kernel(const __bf16* __restrict__ XB,   // [4096][512]
                     const __bf16* __restrict__ WQ,   // [1536][512]
                     __bf16* __restrict__ Qb,         // [B,H,L,DH]
                     __bf16* __restrict__ Kb,         // [B,H,L,DH]
                     __bf16* __restrict__ Vt)         // [B,H,DH,L]
{
    const int lane = threadIdx.x & 31;
    const int wave = threadIdx.x >> 5;
    const int g = lane >> 4;
    const int c = lane & 15;
    const int m0 = blockIdx.x * 128 + wave * 16;
    const int n0 = blockIdx.y * 64;

    v8f acc[4] = {};
    for (int k0 = 0; k0 < D_; k0 += 32) {
        v16bf a = load_a_frag(XB, D_, m0, k0, lane);
#pragma unroll
        for (int t = 0; t < 4; ++t) {
            v16bf bf = load_b_frag(WQ, D_, n0 + 16 * t, k0, lane);
            acc[t] = __builtin_amdgcn_wmma_f32_16x16x32_bf16(false, a, false, bf,
                                                             (short)0, acc[t], false, false);
        }
    }

    const int sec = n0 / 512;           // 0=Q, 1=K, 2=V (64-wide n-tiles never straddle)
    const int h   = (n0 % 512) / 64;
#pragma unroll
    for (int t = 0; t < 4; ++t) {
        const int dp = 16 * t + c;
#pragma unroll
        for (int r = 0; r < 8; ++r) {
            const int m = m0 + r + 8 * g;
            const int b = m >> 11;
            const int l = m & (L_ - 1);
            const __bf16 val = (__bf16)acc[t][r];
            const size_t bh = (size_t)b * H_ + h;
            if (sec == 0)      Qb[(bh * L_ + l) * DH_ + dp] = val;
            else if (sec == 1) Kb[(bh * L_ + l) * DH_ + dp] = val;
            else               Vt[(bh * DH_ + dp) * L_ + l] = val;
        }
    }
}

// Flash attention with Gaussian distance bias.
// 4 waves/block, wave owns 16 query rows; K/V tiles staged to LDS by TDM, double-buffered.
__global__ __launch_bounds__(128)
void attn_flash_kernel(const __bf16* __restrict__ Qb,
                       const __bf16* __restrict__ Kb,
                       const __bf16* __restrict__ Vt,
                       const float* __restrict__ positions,
                       const float* __restrict__ log_sigma,
                       const float* __restrict__ gbias_p,
                       const int* __restrict__ ns_p,
                       __bf16* __restrict__ Ob)   // [B*L][H*DH]
{
    __shared__ alignas(16) __bf16 Ksh[2][64][KPAD];   // TDM destination (padded rows)
    __shared__ alignas(16) __bf16 Vsh[2][64][KPAD];   // rows = d, cols = local j
    __shared__ alignas(16) float  Ssh[4][16][68];
    __shared__ alignas(32) __bf16 Psh[4][16][64];
    __shared__ float Ash[4][16];
    __shared__ float Lsh[4][16];

    const int lane = threadIdx.x & 31;
    const int wave = threadIdx.x >> 5;
    const int g = lane >> 4;
    const int c = lane & 15;

    const int bh = blockIdx.y;
    const int b  = bh >> 3;
    const int h  = bh & 7;
    const int i0 = blockIdx.x * 64 + wave * 16;

    const int   Ls     = *ns_p;
    const float gbias  = *gbias_p;
    const float sig    = __expf(log_sigma[h]);
    const float inv2s2 = 0.5f / (sig * sig);
    const float scale  = 0.125f;       // 1/sqrt(64)

    const __bf16* Qh = Qb + (size_t)bh * L_ * DH_;
    const __bf16* Kh = Kb + (size_t)bh * L_ * DH_;
    const __bf16* Vh = Vt + (size_t)bh * DH_ * L_;
    const float*  pos = positions + (size_t)b * Ls * 2;

    const unsigned koff[2] = { (unsigned)(size_t)&Ksh[0][0][0],
                               (unsigned)(size_t)&Ksh[1][0][0] };
    const unsigned voff[2] = { (unsigned)(size_t)&Vsh[0][0][0],
                               (unsigned)(size_t)&Vsh[1][0][0] };

    // Q fragments for this wave's 16 rows (K = 64 -> 2 frags), reused all iterations
    v16bf qa0 = load_a_frag(Qh, DH_, i0, 0,  lane);
    v16bf qa1 = load_a_frag(Qh, DH_, i0, 32, lane);

    // per-lane row positions for the C-fragment rows r + 8*g
    float pix[8], piy[8];
    bool  iok[8];
#pragma unroll
    for (int r = 0; r < 8; ++r) {
        const int i = i0 + r + 8 * g;
        iok[r] = (i < Ls);
        pix[r] = iok[r] ? pos[(size_t)i * 2 + 0] : 0.0f;
        piy[r] = iok[r] ? pos[(size_t)i * 2 + 1] : 0.0f;
    }

    // prologue: TDM-stage tile 0 into buffer 0
    if (wave == 0) {
        tdm_load_2d(koff[0], Kh, DH_, L_, DH_, 64, 64);          // K rows j0..j0+63
        tdm_load_2d(voff[0], Vh, L_, DH_, L_, 64, 64);           // V cols j0..j0+63 (d-major)
        __builtin_amdgcn_s_wait_tensorcnt(0);
    }
    __syncthreads();

    float m_row = -1.0e30f;            // lanes 0..15 own one row each
    float l_row = 0.0f;
    v8f oacc[4] = {};

    const int NTILE = L_ / 64;
    for (int it = 0; it < NTILE; ++it) {
        const int j0  = it * 64;
        const int cur = it & 1;
        // kick off next tile's DMA into the other buffer
        if (wave == 0 && it + 1 < NTILE) {
            const int nj = j0 + 64;
            tdm_load_2d(koff[cur ^ 1], Kh + (size_t)nj * DH_, DH_, L_, DH_, 64, 64);
            tdm_load_2d(voff[cur ^ 1], Vh + nj,               L_, DH_, L_, 64, 64);
        }
        const __bf16* kp = &Ksh[cur][0][0];
        const __bf16* vp = &Vsh[cur][0][0];

        // ---- phase 1: S = Q K^T * scale + bias -> LDS (wave-private slice)
#pragma unroll
        for (int t = 0; t < 4; ++t) {
            v16bf kb0 = load_b_frag_lds(kp, KPAD, 16 * t, 0,  lane);
            v16bf kb1 = load_b_frag_lds(kp, KPAD, 16 * t, 32, lane);
            v8f s = {};
            s = __builtin_amdgcn_wmma_f32_16x16x32_bf16(false, qa0, false, kb0, (short)0, s, false, false);
            s = __builtin_amdgcn_wmma_f32_16x16x32_bf16(false, qa1, false, kb1, (short)0, s, false, false);
            const int j = j0 + 16 * t + c;
            const bool jok = (j < Ls);
            const float pjx = jok ? pos[(size_t)j * 2 + 0] : 0.0f;
            const float pjy = jok ? pos[(size_t)j * 2 + 1] : 0.0f;
#pragma unroll
            for (int r = 0; r < 8; ++r) {
                float bias = gbias;
                if (jok && iok[r]) {
                    const float dx = pix[r] - pjx;
                    const float dy = piy[r] - pjy;
                    bias = -(dx * dx + dy * dy) * inv2s2;
                }
                Ssh[wave][r + 8 * g][16 * t + c] = s[r] * scale + bias;
            }
        }
        __syncthreads();

        // ---- phase 2: online softmax, row per lane (0..15); P -> LDS as bf16
        if (lane < 16) {
            float tmax = -1.0e30f;
#pragma unroll 8
            for (int j = 0; j < 64; ++j) tmax = fmaxf(tmax, Ssh[wave][lane][j]);
            const float mnew  = fmaxf(m_row, tmax);
            const float alpha = __expf(m_row - mnew);
            float psum = 0.0f;
#pragma unroll 8
            for (int j = 0; j < 64; ++j) {
                const float p = __expf(Ssh[wave][lane][j] - mnew);
                psum += p;
                Psh[wave][lane][j] = (__bf16)p;
            }
            l_row = l_row * alpha + psum;
            m_row = mnew;
            Ash[wave][lane] = alpha;
        }
        __syncthreads();

        // ---- phase 3: O = diag(alpha)*O + P @ V
        float arow[8];
#pragma unroll
        for (int r = 0; r < 8; ++r) arow[r] = Ash[wave][r + 8 * g];
        const __bf16* pp = &Psh[wave][0][0];
        v16bf pa0 = load_a_frag(pp, 64, 0, 0,  lane);
        v16bf pa1 = load_a_frag(pp, 64, 0, 32, lane);
#pragma unroll
        for (int t = 0; t < 4; ++t) {
#pragma unroll
            for (int r = 0; r < 8; ++r) oacc[t][r] *= arow[r];
            v16bf vb0 = load_b_frag_lds(vp, KPAD, 16 * t, 0,  lane);
            v16bf vb1 = load_b_frag_lds(vp, KPAD, 16 * t, 32, lane);
            oacc[t] = __builtin_amdgcn_wmma_f32_16x16x32_bf16(false, pa0, false, vb0, (short)0, oacc[t], false, false);
            oacc[t] = __builtin_amdgcn_wmma_f32_16x16x32_bf16(false, pa1, false, vb1, (short)0, oacc[t], false, false);
        }
        // next tile's DMA must be complete before buffers swap
        if (wave == 0 && it + 1 < NTILE) __builtin_amdgcn_s_wait_tensorcnt(0);
        __syncthreads();
    }

    // ---- epilogue: normalize by 1/l, emit O as [B*L][H*DH] bf16
    if (lane < 16) Lsh[wave][lane] = 1.0f / l_row;
    __syncthreads();
    float lrec[8];
#pragma unroll
    for (int r = 0; r < 8; ++r) lrec[r] = Lsh[wave][r + 8 * g];
#pragma unroll
    for (int t = 0; t < 4; ++t) {
        const int e = h * DH_ + 16 * t + c;
#pragma unroll
        for (int r = 0; r < 8; ++r) {
            const int i = i0 + r + 8 * g;
            Ob[((size_t)b * L_ + i) * (H_ * DH_) + e] = (__bf16)(oacc[t][r] * lrec[r]);
        }
    }
}

// out = O @ w_out^T + b_out ; M=4096, N=512, K=512, f32 output
__global__ __launch_bounds__(256)
void out_gemm_kernel(const __bf16* __restrict__ Ob,   // [4096][512]
                     const __bf16* __restrict__ WO,   // [512][512]
                     const float* __restrict__ bout,
                     float* __restrict__ out)         // [4096][512]
{
    const int lane = threadIdx.x & 31;
    const int wave = threadIdx.x >> 5;
    const int g = lane >> 4;
    const int c = lane & 15;
    const int m0 = blockIdx.x * 128 + wave * 16;
    const int n0 = blockIdx.y * 64;

    v8f acc[4] = {};
    for (int k0 = 0; k0 < 512; k0 += 32) {
        v16bf a = load_a_frag(Ob, 512, m0, k0, lane);
#pragma unroll
        for (int t = 0; t < 4; ++t) {
            v16bf bf = load_b_frag(WO, 512, n0 + 16 * t, k0, lane);
            acc[t] = __builtin_amdgcn_wmma_f32_16x16x32_bf16(false, a, false, bf,
                                                             (short)0, acc[t], false, false);
        }
    }
#pragma unroll
    for (int t = 0; t < 4; ++t) {
        const int n = n0 + 16 * t + c;
        const float bo = bout[n];
#pragma unroll
        for (int r = 0; r < 8; ++r) {
            const int m = m0 + r + 8 * g;
            out[(size_t)m * 512 + n] = acc[t][r] + bo;
        }
    }
}

// ---------------------------------------------------------------- launch

extern "C" void kernel_launch(void* const* d_in, const int* in_sizes, int n_in,
                              void* d_out, int out_size, void* d_ws, size_t ws_size,
                              hipStream_t stream) {
    (void)in_sizes; (void)n_in; (void)out_size; (void)ws_size;

    const float* x         = (const float*)d_in[0];   // [2,2048,512]
    const float* positions = (const float*)d_in[1];   // [2,Ls,2]
    const float* w_qkv     = (const float*)d_in[2];   // [1536,512]
    const float* w_out     = (const float*)d_in[3];   // [512,512]
    const float* b_out     = (const float*)d_in[4];   // [512]
    const float* log_sigma = (const float*)d_in[5];   // [8]
    const float* gbias     = (const float*)d_in[6];   // scalar
    const int*   nspatial  = (const int*)d_in[7];     // scalar

    char* ws = (char*)d_ws;
    __bf16* XB   = (__bf16*)(ws + 0);          // 4096*512   bf16 = 4 MB
    __bf16* WQKV = (__bf16*)(ws + 4194304);    // 1536*512   bf16 = 1.5 MB
    __bf16* WOB  = (__bf16*)(ws + 5767168);    // 512*512    bf16 = 0.5 MB
    __bf16* Qb   = (__bf16*)(ws + 6291456);    // 2*8*2048*64 bf16 = 4 MB
    __bf16* Kb   = (__bf16*)(ws + 10485760);   // 4 MB
    __bf16* Vt   = (__bf16*)(ws + 14680064);   // 4 MB
    __bf16* Ob   = (__bf16*)(ws + 18874368);   // 4096*512   bf16 = 4 MB

    cvt_f32_to_bf16_kernel<<<(2097152 + 255) / 256, 256, 0, stream>>>(x, XB, 2097152);
    cvt_f32_to_bf16_kernel<<<(786432 + 255) / 256, 256, 0, stream>>>(w_qkv, WQKV, 786432);
    cvt_f32_to_bf16_kernel<<<(262144 + 255) / 256, 256, 0, stream>>>(w_out, WOB, 262144);

    qkv_gemm_kernel<<<dim3(M_ / 128, E3_ / 64), 256, 0, stream>>>(XB, WQKV, Qb, Kb, Vt);

    attn_flash_kernel<<<dim3(L_ / 64, B_ * H_), 128, 0, stream>>>(
        Qb, Kb, Vt, positions, log_sigma, gbias, nspatial, Ob);

    out_gemm_kernel<<<dim3(M_ / 128, D_ / 64), 256, 0, stream>>>(Ob, WOB, b_out, (float*)d_out);
}
